// TokenFoam_47811575939408
// MI455X (gfx1250) — compile-verified, matrix operand order checked by reading, not run on hardware
//
#include <hip/hip_runtime.h>
#include <hip/hip_bf16.h>

#define VOCAB 32000
#define DIM   256
#define NBUB  32
#define NEQS  5
#define SEQL  64
#define MTILE 128   // 8 waves x 16-row M-tiles per block

typedef __bf16 bf16x8  __attribute__((ext_vector_type(8)));
typedef __bf16 v16bf   __attribute__((ext_vector_type(16)));
typedef float  v8f     __attribute__((ext_vector_type(8)));
typedef int    v4i_g   __attribute__((vector_size(16)));   // GCC-style, matches builtin proto

#define AS_GLOBAL __attribute__((address_space(1)))
#define AS_LDS    __attribute__((address_space(3)))

#if defined(__has_builtin)
# if __has_builtin(__builtin_amdgcn_global_load_async_to_lds_b128) && \
     __has_builtin(__builtin_amdgcn_s_wait_asynccnt)
#  define HAVE_ASYNC_LDS 1
# endif
#endif
#ifndef HAVE_ASYNC_LDS
# define HAVE_ASYNC_LDS 0
#endif

static __device__ inline __bf16 f2bf(float f) {
    union { float f; unsigned u; } in; in.f = f;
    unsigned u = in.u;
    unsigned r = u + 0x7FFFu + ((u >> 16) & 1u);   // round-to-nearest-even
    union { unsigned short s; __bf16 b; } out;
    out.s = (unsigned short)(r >> 16);
    return out.b;
}

// ---------------------------------------------------------------------------
// Kernel 0: convert E (f32) -> bf16 once; E_bf16 then lives in the 192MB L2.
// ---------------------------------------------------------------------------
__global__ void convert_e_kernel(const float* __restrict__ E,
                                 __bf16* __restrict__ Eb, int n) {
    int i = blockIdx.x * blockDim.x + threadIdx.x;
    if (i < n) Eb[i] = f2bf(E[i]);
}

// ---------------------------------------------------------------------------
// Kernel 1: the sequential 64-token scan (memory dynamics + foam relaxation).
// One workgroup (256 threads = 8 waves, wave32). Emits per step:
//   sn_all[step][b][d]  (bf16, normalized bubble states)
//   p_all[step][b]      (f32, output distribution)
// memory state [32][256] kept in workspace global (zeroed at entry).
// ---------------------------------------------------------------------------
__global__ __launch_bounds__(256) void foam_scan_kernel(
    const int*   __restrict__ tokens,
    const float* __restrict__ E,
    const float* __restrict__ bubbles,
    const float* __restrict__ pbase,
    const float* __restrict__ psens,
    float*  __restrict__ mem_g,     // [NBUB][DIM]
    __bf16* __restrict__ sn_all,    // [SEQL][NBUB][DIM]
    float*  __restrict__ p_all)     // [SEQL][NBUB]
{
    __shared__ float s_mat[NBUB][DIM + 1];   // padded: column access conflict-free
    __shared__ float xwm[DIM];
    __shared__ float part[NBUB][8];
    __shared__ float wvec[NBUB];
    __shared__ float redA[8], redB[8], redC[8];
    __shared__ float scal[1];

    const int t    = threadIdx.x;        // == dimension index d (0..255)
    const int lane = t & 31;
    const int wv   = t >> 5;
    const float base = *pbase;
    const float sens = fabsf(*psens);
    const float scale = 0.0625f;         // 1/sqrt(256)

    // zero memory state
    for (int i = t; i < NBUB * DIM; i += 256) mem_g[i] = 0.f;
    __syncthreads();

    for (int step = 0; step < SEQL; ++step) {
        const int tok = tokens[step];
        const float xd = E[(size_t)tok * DIM + t];

        // mm = memory.mean(axis=0), per-dimension
        float mmd = 0.f;
        for (int r = 0; r < NBUB; ++r) mmd += mem_g[r * DIM + t];
        mmd *= (1.f / NBUB);

        // block reductions: ||x||^2, ||mm||^2, x.mm
        float a = xd * xd, b2 = mmd * mmd, c = xd * mmd;
        for (int off = 16; off; off >>= 1) {
            a  += __shfl_xor(a,  off);
            b2 += __shfl_xor(b2, off);
            c  += __shfl_xor(c,  off);
        }
        if (lane == 0) { redA[wv] = a; redB[wv] = b2; redC[wv] = c; }
        __syncthreads();
        if (t == 0) {
            float A = 0.f, B = 0.f, C = 0.f;
            for (int w = 0; w < 8; ++w) { A += redA[w]; B += redB[w]; C += redC[w]; }
            const float xn = sqrtf(A) + 1e-10f;
            const float mn = sqrtf(B) + 1e-10f;
            const float novelty = (mn > 1e-8f) ? (1.f - C / (xn * mn)) : 1.f;
            scal[0] = 1.f / (1.f + expf(-(base - sens * novelty)));  // sigmoid
        }
        __syncthreads();
        const float decay = scal[0];
        xwm[t] = xd + decay * mmd;

        // init foam state from bubbles
        for (int r = 0; r < NBUB; ++r) s_mat[r][t] = bubbles[r * DIM + t];
        __syncthreads();

        // NEQS relaxation steps + one final softmax (=p)
        for (int it = 0; it <= NEQS; ++it) {
            // dots: 32 rows x 8 partial segments of 32 dims
            {
                const int r = t >> 3, seg = t & 7, d0 = seg * 32;
                float ps = 0.f;
                for (int d = d0; d < d0 + 32; ++d) ps += s_mat[r][d] * xwm[d];
                part[r][seg] = ps;
            }
            __syncthreads();
            if (t < NBUB) {          // wave 0: 32-wide softmax via shuffles
                float dot = 0.f;
                for (int g = 0; g < 8; ++g) dot += part[t][g];
                dot *= scale;
                float m = dot;
                for (int off = 16; off; off >>= 1) m = fmaxf(m, __shfl_xor(m, off));
                float e = expf(dot - m);
                float ssum = e;
                for (int off = 16; off; off >>= 1) ssum += __shfl_xor(ssum, off);
                wvec[t] = e / ssum;
            }
            __syncthreads();
            if (it < NEQS) {         // relax toward x_wm
                const float xv = xwm[t];
                for (int r = 0; r < NBUB; ++r) {
                    const float sv = s_mat[r][t];
                    s_mat[r][t] = sv + wvec[r] * (xv - sv);
                }
                __syncthreads();
            }
        }
        // wvec now holds p
        if (t < NBUB) p_all[step * NBUB + t] = wvec[t];
        __syncthreads();

        // row inverse norms
        {
            const int r = t >> 3, seg = t & 7, d0 = seg * 32;
            float ps = 0.f;
            for (int d = d0; d < d0 + 32; ++d) { const float v = s_mat[r][d]; ps += v * v; }
            part[r][seg] = ps;
        }
        __syncthreads();
        if (t < NBUB) {
            float nrm = 0.f;
            for (int g = 0; g < 8; ++g) nrm += part[t][g];
            wvec[t] = 1.f / (sqrtf(nrm) + 1e-10f);
        }
        __syncthreads();

        // emit sn (bf16) and update memory state
        for (int r = 0; r < NBUB; ++r) {
            const float sv = s_mat[r][t];
            sn_all[((size_t)step * NBUB + r) * DIM + t] = f2bf(sv * wvec[r]);
            mem_g[r * DIM + t] = decay * mem_g[r * DIM + t] + (1.f - decay) * sv;
        }
        __syncthreads();
    }
}

// ---------------------------------------------------------------------------
// Kernel 2: Born-rule logits via WMMA.
//   G = E_bf16[MTILE x 256] x sn_step^T[256 x 32]   (v_wmma_f32_16x16x32_bf16)
//   logits[m, step] = sum_b p[b] * G[m][b]^2
// 250 blocks x 256 threads (8 waves). Each wave owns one 16-row M-tile whose
// A fragments (8 x v16bf = 64 VGPRs) are hoisted into registers ONCE and
// reused across all 64 steps; the inner loop only streams B (sn) from LDS.
// sn tiles are staged with GLOBAL_LOAD_ASYNC_TO_LDS (ASYNCcnt) when the
// builtin exists, else a plain vectorized copy.
// ---------------------------------------------------------------------------
__global__ __launch_bounds__(256) void born_logits_kernel(
    const __bf16* __restrict__ Eb,      // [VOCAB][DIM]
    const __bf16* __restrict__ sn_all,  // [SEQL][NBUB][DIM]
    const float*  __restrict__ p_all,   // [SEQL][NBUB]
    float*        __restrict__ out)     // [SEQL][VOCAB]
{
    __shared__ __bf16 sn_tile[NBUB * DIM];   // 16 KB
    __shared__ float  p_sh[NBUB];

    const int t    = threadIdx.x;
    const int lane = t & 31;
    const int wave = t >> 5;        // m-tile 0..7
    const int m0   = blockIdx.x * MTILE;
    const int nl   = lane & 15;     // N within tile / A-row within tile
    const int kh   = lane >> 4;     // K-half selector

    // Hoist this wave's A fragments (rows of E, K=256) into registers.
    // 16-bit A layout: lane half kh holds K = {kh*8..kh*8+7} in V0-3 and
    // K = {16+kh*8..} in V4-7 within each 32-wide K block.
    v16bf afr[8];
    {
        const __bf16* arow = Eb + (size_t)(m0 + wave * 16 + nl) * DIM;
#pragma unroll
        for (int kb8 = 0; kb8 < 8; ++kb8) {
            bf16x8 lo = *(const bf16x8*)(arow + kb8 * 32 + kh * 8);
            bf16x8 hi = *(const bf16x8*)(arow + kb8 * 32 + 16 + kh * 8);
            afr[kb8] = __builtin_shufflevector(lo, hi,
                0, 1, 2, 3, 4, 5, 6, 7, 8, 9, 10, 11, 12, 13, 14, 15);
        }
    }

    const __bf16* b0row = sn_tile + nl * DIM;          // B: bubble nl
    const __bf16* b1row = sn_tile + (16 + nl) * DIM;   // B: bubble 16+nl

    for (int step = 0; step < SEQL; ++step) {
        __syncthreads();   // guards sn_tile reuse across steps
        {
            const __bf16* src = sn_all + (size_t)step * NBUB * DIM;
#if HAVE_ASYNC_LDS
            for (int i = t; i < NBUB * DIM / 8; i += 256) {
                __builtin_amdgcn_global_load_async_to_lds_b128(
                    (AS_GLOBAL v4i_g*)(src + i * 8),
                    (AS_LDS    v4i_g*)(sn_tile + i * 8),
                    0, 0);
            }
            if (t < NBUB) p_sh[t] = p_all[step * NBUB + t];
            __builtin_amdgcn_s_wait_asynccnt(0);
#else
            const uint4* s4 = (const uint4*)src;
            uint4* dst = (uint4*)sn_tile;
            for (int i = t; i < NBUB * DIM / 8; i += 256) dst[i] = s4[i];
            if (t < NBUB) p_sh[t] = p_all[step * NBUB + t];
#endif
        }
        if (step + 1 < SEQL)  // pull next step's sn tile toward this WGP
            __builtin_prefetch(sn_all + (size_t)(step + 1) * NBUB * DIM, 0, 1);
        __syncthreads();

        v8f c0 = {0.f, 0.f, 0.f, 0.f, 0.f, 0.f, 0.f, 0.f};
        v8f c1 = {0.f, 0.f, 0.f, 0.f, 0.f, 0.f, 0.f, 0.f};

#pragma unroll
        for (int kb8 = 0; kb8 < 8; ++kb8) {
            const int o_lo = kb8 * 32 + kh * 8;
            const int o_hi = o_lo + 16;

            bf16x8 b0lo = *(const bf16x8*)(b0row + o_lo);
            bf16x8 b0hi = *(const bf16x8*)(b0row + o_hi);
            v16bf b0 = __builtin_shufflevector(b0lo, b0hi,
                0, 1, 2, 3, 4, 5, 6, 7, 8, 9, 10, 11, 12, 13, 14, 15);

            bf16x8 b1lo = *(const bf16x8*)(b1row + o_lo);
            bf16x8 b1hi = *(const bf16x8*)(b1row + o_hi);
            v16bf b1 = __builtin_shufflevector(b1lo, b1hi,
                0, 1, 2, 3, 4, 5, 6, 7, 8, 9, 10, 11, 12, 13, 14, 15);

            c0 = __builtin_amdgcn_wmma_f32_16x16x32_bf16(
                     false, afr[kb8], false, b0, (short)0, c0, false, false);
            c1 = __builtin_amdgcn_wmma_f32_16x16x32_bf16(
                     false, afr[kb8], false, b1, (short)0, c1, false, false);
        }

        // Born reduce: each lane has G[m][nl] (bubbles nl and 16+nl) for 8 rows.
        const float pb0 = p_sh[nl];
        const float pb1 = p_sh[16 + nl];
        float* outrow = out + (size_t)step * VOCAB + m0 + wave * 16 + kh * 8;
#pragma unroll
        for (int i = 0; i < 8; ++i) {
            float v = pb0 * c0[i] * c0[i] + pb1 * c1[i] * c1[i];
            v += __shfl_xor(v, 1);
            v += __shfl_xor(v, 2);
            v += __shfl_xor(v, 4);
            v += __shfl_xor(v, 8);     // sum over all 32 bubbles (16 lanes x 2) in this half
            if (nl == 0) outrow[i] = v;   // row m = m0 + wave*16 + kh*8 + i
        }
    }
}

// ---------------------------------------------------------------------------
extern "C" void kernel_launch(void* const* d_in, const int* in_sizes, int n_in,
                              void* d_out, int out_size, void* d_ws, size_t ws_size,
                              hipStream_t stream) {
    const int*   tokens  = (const int*)  d_in[0];
    const float* E       = (const float*)d_in[1];
    const float* bubbles = (const float*)d_in[2];
    const float* pbase   = (const float*)d_in[3];
    const float* psens   = (const float*)d_in[4];
    float* out = (float*)d_out;

    // workspace layout (all 16B aligned)
    char* ws = (char*)d_ws;
    __bf16* Eb     = (__bf16*)(ws);                                  // 16,384,000 B
    __bf16* sn_all = (__bf16*)(ws + 16384000);                       //  1,048,576 B
    float*  p_all  = (float*) (ws + 16384000 + 1048576);             //      8,192 B
    float*  mem_g  = (float*) (ws + 16384000 + 1048576 + 8192);      //     32,768 B
    (void)in_sizes; (void)n_in; (void)out_size; (void)ws_size;

    const int nE = VOCAB * DIM;
    convert_e_kernel<<<(nE + 255) / 256, 256, 0, stream>>>(E, Eb, nE);

    foam_scan_kernel<<<1, 256, 0, stream>>>(tokens, E, bubbles, pbase, psens,
                                            mem_g, sn_all, p_all);

    born_logits_kernel<<<VOCAB / MTILE, 256, 0, stream>>>(Eb, sn_all, p_all, out);
}
